// RecurrentMoE_82231443849809
// MI455X (gfx1250) — compile-verified, matrix operand order checked by reading, not run on hardware
//
#include <hip/hip_runtime.h>
#include <hip/hip_bf16.h>
#include <math.h>

#define B_  2
#define S_  2048
#define D_  1024
#define H_  4096
#define E_  8
#define RH_ 256
#define G3_ (3*RH_)   // 768
#define NT_ (B_*S_)   // 4096 tokens

typedef __attribute__((ext_vector_type(16))) __bf16 v16bf;
typedef __attribute__((ext_vector_type(8)))  float  v8f;

union Frag { unsigned int u[8]; v16bf v; };

// ---- Tensor Data Mover availability (guarded: falls back to sync loads) ----
#if defined(__has_builtin)
#  if __has_builtin(__builtin_amdgcn_tensor_load_to_lds) && \
      __has_builtin(__builtin_amdgcn_s_wait_tensorcnt)
#    define USE_TDM 1
#  endif
#endif
#ifndef USE_TDM
#  define USE_TDM 0
#endif

#if USE_TDM
typedef __attribute__((ext_vector_type(4))) unsigned int u32x4_t;
typedef __attribute__((ext_vector_type(8))) int          i32x8_t;
typedef __attribute__((ext_vector_type(4))) int          i32x4_t;

// Issue one TDM load of a 64x64 bf16 tile (row-major, K elements per tensor row)
// into LDS at byte offset lds_off, padding each 128B tile row with 16B so the
// LDS stride matches the [64][72] ushort tiles (pad_interval=32 dwords (code 4),
// pad_amount=4 dwords (code 3)).  D# layout per CDNA5 ISA 08_async_tensor §8.
__device__ __forceinline__ void tdm_tile_load(const unsigned short* gsrc,
                                              unsigned int lds_off,
                                              int K, int M) {
  unsigned long long ga = (unsigned long long)(const void*)gsrc;
  u32x4_t g0 = { 1u,                                   // count=1 (valid D#)
                 lds_off,                               // lds_addr
                 (unsigned int)ga,                      // global_addr[31:0]
                 (unsigned int)((ga >> 32) & 0x01FFFFFFull) | 0x80000000u }; // type=2
  i32x8_t g1 = { (int)((1u << 16) | (1u << 20) | (4u << 22) | (3u << 25)),
                 //    data_size=2B  pad_en      pad_ivl=32dw  pad_amt=4dw
                 (int)(((unsigned)K & 0xFFFFu) << 16),          // tensor_dim0 lo16
                 (int)((((unsigned)K >> 16) & 0xFFFFu) |        // tensor_dim0 hi16
                       (((unsigned)M & 0xFFFFu) << 16)),        // tensor_dim1 lo16
                 (int)((((unsigned)M >> 16) & 0xFFFFu) |        // tensor_dim1 hi16
                       (64u << 16)),                            // tile_dim0 = 64
                 64,                                            // tile_dim1 = 64
                 K,                                             // dim0_stride lo32
                 (int)(((unsigned)K >> 16) & 0xFFFFu),          // dim0_stride hi16
                 0 };
  i32x4_t z4 = { 0, 0, 0, 0 };
#if __clang_major__ >= 23
  i32x8_t z8 = { 0, 0, 0, 0, 0, 0, 0, 0 };
  __builtin_amdgcn_tensor_load_to_lds(g0, g1, z4, z4, z8, 0);
#else
  __builtin_amdgcn_tensor_load_to_lds(g0, g1, z4, z4, 0);
#endif
}
#endif

__device__ __forceinline__ unsigned short f2bf(float f) {
  unsigned int u = __float_as_uint(f);
  u += 0x7FFFu + ((u >> 16) & 1u);           // round-to-nearest-even
  return (unsigned short)(u >> 16);
}
__device__ __forceinline__ unsigned int pack2bf(float a, float b) {
  return (unsigned int)f2bf(a) | ((unsigned int)f2bf(b) << 16);
}

// Load one 16x32 bf16 A/B fragment from an LDS tile row (K contiguous).
// Per ISA layout: lane = {hi,ln}; VGPR v holds K pair (2v | 16+2(v-4)) + 8*hi.
__device__ __forceinline__ v16bf load_frag(const unsigned short* row, int kk, int hi) {
  Frag f;
#pragma unroll
  for (int v = 0; v < 8; ++v) {
    int k = kk + ((v < 4) ? 2 * v : 16 + 2 * (v - 4)) + 8 * hi;
    f.u[v] = *(const unsigned int*)(row + k);
  }
  return f.v;
}

__device__ __forceinline__ v8f wmma_bf16(v16bf a, v16bf b, v8f c) {
  return __builtin_amdgcn_wmma_f32_16x16x32_bf16(false, a, false, b, (short)0, c,
                                                 false, false);
}

// ---------------------------------------------------------------- utilities
__global__ void k_zero(float* p, int n) {
  int i = threadIdx.x;
  if (i < n) p[i] = 0.f;
}

__global__ void k_cvt_bf16(const float* __restrict__ x,
                           unsigned short* __restrict__ y, int n) {
  int i = blockIdx.x * blockDim.x + threadIdx.x;
  int stride = gridDim.x * blockDim.x;
  for (; i < n / 4; i += stride) {
    float4 f = ((const float4*)x)[i];
    uint2 p;
    p.x = pack2bf(f.x, f.y);
    p.y = pack2bf(f.z, f.w);
    ((uint2*)y)[i] = p;
  }
}

// ------------------------------------------------- C = A(bf16) @ W(f32)^T + bias
// Tile: 64(M) x 128(N), BK=64, 256 threads = 8 waves, 32x32 per wave.
__global__ __launch_bounds__(256) void k_gemm_bias(
    const unsigned short* __restrict__ A, const float* __restrict__ W,
    const float* __restrict__ bias, float* __restrict__ C, int M, int N, int K) {
  __shared__ unsigned short As[64][72];
  __shared__ unsigned short Ws[128][72];
  int tid = threadIdx.x;
  int lane = tid & 31, wave = tid >> 5;
  int wm = wave >> 2, wn = wave & 3;
  int hi = lane >> 4, ln = lane & 15;
  int mBase = blockIdx.x * 64, nBase = blockIdx.y * 128;
#if USE_TDM
  unsigned int ldsA = (unsigned int)(unsigned long long)(const void*)&As[0][0];
#endif

  v8f acc[4] = {};
  for (int k0 = 0; k0 < K; k0 += 64) {
#if USE_TDM
    if (tid == 0)                             // TDM: A tile 64x64 bf16 -> LDS
      tdm_tile_load(A + (size_t)mBase * K + k0, ldsA, K, M);
#else
#pragma unroll
    for (int i = 0; i < 2; ++i) {             // A tile: 64x64 bf16
      int g = tid + 256 * i;
      int r = g >> 3, c = (g & 7) * 8;
      *(uint4*)&As[r][c] = *(const uint4*)(A + (size_t)(mBase + r) * K + k0 + c);
    }
#endif
#pragma unroll
    for (int i = 0; i < 8; ++i) {             // W tile: 128x64 f32 -> bf16
      int g = tid + 256 * i;
      int r = g >> 4, c = (g & 15) * 4;
      float4 f = *(const float4*)(W + (size_t)(nBase + r) * K + k0 + c);
      uint2 p;
      p.x = pack2bf(f.x, f.y);
      p.y = pack2bf(f.z, f.w);
      *(uint2*)&Ws[r][c] = p;
    }
    if (k0 + 64 < K)                          // global_prefetch_b8 of next tile
      __builtin_prefetch(W + (size_t)(nBase + (tid >> 1)) * K + k0 + 64, 0, 0);
#if USE_TDM
    __builtin_amdgcn_s_wait_tensorcnt(0);
#endif
    __syncthreads();
#pragma unroll
    for (int kk = 0; kk < 64; kk += 32) {
      v16bf a0 = load_frag(&As[wm * 32 + ln][0], kk, hi);
      v16bf a1 = load_frag(&As[wm * 32 + 16 + ln][0], kk, hi);
      v16bf b0 = load_frag(&Ws[wn * 32 + ln][0], kk, hi);
      v16bf b1 = load_frag(&Ws[wn * 32 + 16 + ln][0], kk, hi);
      acc[0] = wmma_bf16(a0, b0, acc[0]);
      acc[1] = wmma_bf16(a0, b1, acc[1]);
      acc[2] = wmma_bf16(a1, b0, acc[2]);
      acc[3] = wmma_bf16(a1, b1, acc[3]);
    }
    __syncthreads();
  }
#pragma unroll
  for (int ms = 0; ms < 2; ++ms)
#pragma unroll
    for (int ns = 0; ns < 2; ++ns) {
      int col = nBase + wn * 32 + ns * 16 + ln;
      float bv = bias ? bias[col] : 0.f;
      v8f a = acc[ms * 2 + ns];
#pragma unroll
      for (int r = 0; r < 8; ++r) {
        int row = mBase + wm * 32 + ms * 16 + hi * 8 + r;
        C[(size_t)row * N + col] = a[r] + bv;
      }
    }
}

// ---------------------------------- fused gate/up: hb = bf16(gelu(xWg^T)*(xWu^T)*gate)
__global__ __launch_bounds__(256) void k_gateup(
    const unsigned short* __restrict__ A, const float* __restrict__ Wgp,
    const float* __restrict__ Wup, const float* __restrict__ wdense, int expert,
    unsigned short* __restrict__ hb, int M, int N, int K) {
  __shared__ unsigned short As[64][72];
  __shared__ unsigned short Gs[128][72];
  __shared__ unsigned short Us[128][72];
  int tid = threadIdx.x;
  int lane = tid & 31, wave = tid >> 5;
  int wm = wave >> 2, wn = wave & 3;
  int hi = lane >> 4, ln = lane & 15;
  int mBase = blockIdx.x * 64, nBase = blockIdx.y * 128;
#if USE_TDM
  unsigned int ldsA = (unsigned int)(unsigned long long)(const void*)&As[0][0];
#endif

  v8f accG[4] = {};
  v8f accU[4] = {};
  for (int k0 = 0; k0 < K; k0 += 64) {
#if USE_TDM
    if (tid == 0)
      tdm_tile_load(A + (size_t)mBase * K + k0, ldsA, K, M);
#else
#pragma unroll
    for (int i = 0; i < 2; ++i) {
      int g = tid + 256 * i;
      int r = g >> 3, c = (g & 7) * 8;
      *(uint4*)&As[r][c] = *(const uint4*)(A + (size_t)(mBase + r) * K + k0 + c);
    }
#endif
#pragma unroll
    for (int i = 0; i < 8; ++i) {
      int g = tid + 256 * i;
      int r = g >> 4, c = (g & 15) * 4;
      size_t off = (size_t)(nBase + r) * K + k0 + c;
      float4 fg = *(const float4*)(Wgp + off);
      float4 fu = *(const float4*)(Wup + off);
      uint2 pg, pu;
      pg.x = pack2bf(fg.x, fg.y); pg.y = pack2bf(fg.z, fg.w);
      pu.x = pack2bf(fu.x, fu.y); pu.y = pack2bf(fu.z, fu.w);
      *(uint2*)&Gs[r][c] = pg;
      *(uint2*)&Us[r][c] = pu;
    }
    if (k0 + 64 < K) {
      __builtin_prefetch(Wgp + (size_t)(nBase + (tid >> 1)) * K + k0 + 64, 0, 0);
      __builtin_prefetch(Wup + (size_t)(nBase + (tid >> 1)) * K + k0 + 64, 0, 0);
    }
#if USE_TDM
    __builtin_amdgcn_s_wait_tensorcnt(0);
#endif
    __syncthreads();
#pragma unroll
    for (int kk = 0; kk < 64; kk += 32) {
      v16bf a0 = load_frag(&As[wm * 32 + ln][0], kk, hi);
      v16bf a1 = load_frag(&As[wm * 32 + 16 + ln][0], kk, hi);
      v16bf g0 = load_frag(&Gs[wn * 32 + ln][0], kk, hi);
      v16bf g1 = load_frag(&Gs[wn * 32 + 16 + ln][0], kk, hi);
      v16bf u0 = load_frag(&Us[wn * 32 + ln][0], kk, hi);
      v16bf u1 = load_frag(&Us[wn * 32 + 16 + ln][0], kk, hi);
      accG[0] = wmma_bf16(a0, g0, accG[0]);
      accG[1] = wmma_bf16(a0, g1, accG[1]);
      accG[2] = wmma_bf16(a1, g0, accG[2]);
      accG[3] = wmma_bf16(a1, g1, accG[3]);
      accU[0] = wmma_bf16(a0, u0, accU[0]);
      accU[1] = wmma_bf16(a0, u1, accU[1]);
      accU[2] = wmma_bf16(a1, u0, accU[2]);
      accU[3] = wmma_bf16(a1, u1, accU[3]);
    }
    __syncthreads();
  }
#pragma unroll
  for (int ms = 0; ms < 2; ++ms)
#pragma unroll
    for (int ns = 0; ns < 2; ++ns) {
      int col = nBase + wn * 32 + ns * 16 + ln;
      v8f g = accG[ms * 2 + ns];
      v8f u = accU[ms * 2 + ns];
#pragma unroll
      for (int r = 0; r < 8; ++r) {
        int row = mBase + wm * 32 + ms * 16 + hi * 8 + r;
        float wr = wdense[(size_t)row * E_ + expert];
        float gv = g[r];
        float gl = 0.5f * gv * (1.f + erff(gv * 0.70710678118f)); // exact gelu
        hb[(size_t)row * N + col] = f2bf(gl * u[r] * wr);
      }
    }
}

// ---------------------------------------- out (f32) (+)= hb(bf16) @ Wd(f32)^T
__global__ __launch_bounds__(256) void k_down(
    const unsigned short* __restrict__ A, const float* __restrict__ W,
    float* __restrict__ C, int M, int N, int K, int accumulate) {
  __shared__ unsigned short As[64][72];
  __shared__ unsigned short Ws[128][72];
  int tid = threadIdx.x;
  int lane = tid & 31, wave = tid >> 5;
  int wm = wave >> 2, wn = wave & 3;
  int hi = lane >> 4, ln = lane & 15;
  int mBase = blockIdx.x * 64, nBase = blockIdx.y * 128;
#if USE_TDM
  unsigned int ldsA = (unsigned int)(unsigned long long)(const void*)&As[0][0];
#endif

  v8f acc[4] = {};
  for (int k0 = 0; k0 < K; k0 += 64) {
#if USE_TDM
    if (tid == 0)
      tdm_tile_load(A + (size_t)mBase * K + k0, ldsA, K, M);
#else
#pragma unroll
    for (int i = 0; i < 2; ++i) {
      int g = tid + 256 * i;
      int r = g >> 3, c = (g & 7) * 8;
      *(uint4*)&As[r][c] = *(const uint4*)(A + (size_t)(mBase + r) * K + k0 + c);
    }
#endif
#pragma unroll
    for (int i = 0; i < 8; ++i) {
      int g = tid + 256 * i;
      int r = g >> 4, c = (g & 15) * 4;
      float4 f = *(const float4*)(W + (size_t)(nBase + r) * K + k0 + c);
      uint2 p;
      p.x = pack2bf(f.x, f.y);
      p.y = pack2bf(f.z, f.w);
      *(uint2*)&Ws[r][c] = p;
    }
    if (k0 + 64 < K)
      __builtin_prefetch(W + (size_t)(nBase + (tid >> 1)) * K + k0 + 64, 0, 0);
#if USE_TDM
    __builtin_amdgcn_s_wait_tensorcnt(0);
#endif
    __syncthreads();
#pragma unroll
    for (int kk = 0; kk < 64; kk += 32) {
      v16bf a0 = load_frag(&As[wm * 32 + ln][0], kk, hi);
      v16bf a1 = load_frag(&As[wm * 32 + 16 + ln][0], kk, hi);
      v16bf b0 = load_frag(&Ws[wn * 32 + ln][0], kk, hi);
      v16bf b1 = load_frag(&Ws[wn * 32 + 16 + ln][0], kk, hi);
      acc[0] = wmma_bf16(a0, b0, acc[0]);
      acc[1] = wmma_bf16(a0, b1, acc[1]);
      acc[2] = wmma_bf16(a1, b0, acc[2]);
      acc[3] = wmma_bf16(a1, b1, acc[3]);
    }
    __syncthreads();
  }
#pragma unroll
  for (int ms = 0; ms < 2; ++ms)
#pragma unroll
    for (int ns = 0; ns < 2; ++ns) {
      int col = nBase + wn * 32 + ns * 16 + ln;
      v8f a = acc[ms * 2 + ns];
#pragma unroll
      for (int r = 0; r < 8; ++r) {
        int row = mBase + wm * 32 + ms * 16 + hi * 8 + r;
        size_t idx = (size_t)row * N + col;
        C[idx] = accumulate ? (C[idx] + a[r]) : a[r];
      }
    }
}

// ------------------------------------------------------- sequential GRU scan
__global__ __launch_bounds__(768) void k_gru(
    const float* __restrict__ gi, const float* __restrict__ Whh,
    const float* __restrict__ bhh, float* __restrict__ hs,
    float* __restrict__ newh) {
  __shared__ float h[B_][RH_];
  __shared__ float gh[B_][G3_];
  int tid = threadIdx.x;
  if (tid < B_ * RH_) h[tid >> 8][tid & 255] = 0.f;
  __syncthreads();
  for (int t = 0; t < S_; ++t) {
    {  // phase 1: gh = h @ Whh^T + bhh   (all 768 threads)
      const float* wrow = Whh + (size_t)tid * RH_;
      float a0 = 0.f, a1 = 0.f;
      for (int k = 0; k < RH_; ++k) {
        float w = wrow[k];
        a0 += w * h[0][k];
        a1 += w * h[1][k];
      }
      float bb = bhh[tid];
      gh[0][tid] = a0 + bb;
      gh[1][tid] = a1 + bb;
    }
    __syncthreads();
    if (tid < B_ * RH_) {  // phase 2: gates + state update
      int b = tid >> 8, r = tid & 255;
      const float* git = gi + (size_t)(b * S_ + t) * G3_;
      float ir = git[r], iz = git[RH_ + r], in = git[2 * RH_ + r];
      float rg = 1.f / (1.f + expf(-(ir + gh[b][r])));
      float zg = 1.f / (1.f + expf(-(iz + gh[b][RH_ + r])));
      float ng = tanhf(in + rg * gh[b][2 * RH_ + r]);
      float hv = (1.f - zg) * ng + zg * h[b][r];
      hs[(size_t)(b * S_ + t) * RH_ + r] = hv;
      h[b][r] = hv;
      if (t == S_ - 1) newh[b * RH_ + r] = hv;
    }
    __syncthreads();
  }
}

__global__ void k_logits(const float* __restrict__ hs, const float* __restrict__ Wr,
                         const float* __restrict__ br, float* __restrict__ logits) {
  int idx = blockIdx.x * blockDim.x + threadIdx.x;
  if (idx >= NT_ * E_) return;
  int tok = idx >> 3, e = idx & 7;
  const float* hp = hs + (size_t)tok * RH_;
  const float* wp = Wr + (size_t)e * RH_;
  float acc = 0.f;
  for (int r = 0; r < RH_; ++r) acc += hp[r] * wp[r];
  logits[idx] = acc + br[e];
}

__global__ void k_router(const float* __restrict__ logits, float* __restrict__ wdense,
                         float* __restrict__ pacc) {
  int t = blockIdx.x * blockDim.x + threadIdx.x;
  if (t >= NT_) return;
  float l[E_], p[E_];
  float mx = -1e30f;
  for (int e = 0; e < E_; ++e) { l[e] = logits[t * E_ + e]; mx = fmaxf(mx, l[e]); }
  float s = 0.f;
  for (int e = 0; e < E_; ++e) { p[e] = expf(l[e] - mx); s += p[e]; }
  float inv = 1.f / s;
  for (int e = 0; e < E_; ++e) { p[e] *= inv; atomicAdd(&pacc[e], p[e]); }
  int i1 = 0;
  for (int e = 1; e < E_; ++e) if (p[e] > p[i1]) i1 = e;
  int i2 = -1;
  for (int e = 0; e < E_; ++e) {
    if (e == i1) continue;
    if (i2 < 0 || p[e] > p[i2]) i2 = e;
  }
  float d = p[i1] + p[i2];
  for (int e = 0; e < E_; ++e)
    wdense[t * E_ + e] = (e == i1) ? p[i1] / d : ((e == i2) ? p[i2] / d : 0.f);
}

__global__ void k_aux(const float* __restrict__ pacc, float* __restrict__ aux) {
  float s = 0.f;
  for (int e = 0; e < E_; ++e) {
    float m = pacc[e] / (float)NT_;
    s += m * m;
  }
  *aux = (float)E_ * s;
}

// ------------------------------------------------------------------ launcher
extern "C" void kernel_launch(void* const* d_in, const int* in_sizes, int n_in,
                              void* d_out, int out_size, void* d_ws, size_t ws_size,
                              hipStream_t stream) {
  const float* x    = (const float*)d_in[0];
  const float* Wg   = (const float*)d_in[1];
  const float* Wu   = (const float*)d_in[2];
  const float* Wd   = (const float*)d_in[3];
  const float* W_ih = (const float*)d_in[4];
  const float* W_hh = (const float*)d_in[5];
  const float* b_ih = (const float*)d_in[6];
  const float* b_hh = (const float*)d_in[7];
  const float* Wr   = (const float*)d_in[8];
  const float* br   = (const float*)d_in[9];

  float* out    = (float*)d_out;                        // (B,S,D)
  float* logits = out + (size_t)NT_ * D_;               // (B,S,E)
  float* newh   = logits + (size_t)NT_ * E_;            // (B,RH)
  float* aux    = newh + B_ * RH_;                      // scalar

  char* ws = (char*)d_ws;
  unsigned short* xb = (unsigned short*)ws;                          //  8 MiB
  float* gi     = (float*)(ws + 8388608ull);                         // 12 MiB
  float* hs     = (float*)(ws + 20971520ull);                        //  4 MiB
  float* wdense = (float*)(ws + 25165824ull);                        // 128 KiB
  float* pacc   = (float*)(ws + 25296896ull);                        // 256 B
  unsigned short* hb = (unsigned short*)(ws + 25297152ull);          // 32 MiB

  k_zero<<<1, 32, 0, stream>>>(pacc, E_);
  k_cvt_bf16<<<2048, 256, 0, stream>>>(x, xb, NT_ * D_);

  dim3 g1(NT_ / 64, G3_ / 128);  // (64, 6)
  k_gemm_bias<<<g1, 256, 0, stream>>>(xb, W_ih, b_ih, gi, NT_, G3_, D_);

  k_gru<<<1, 768, 0, stream>>>(gi, W_hh, b_hh, hs, newh);
  k_logits<<<(NT_ * E_) / 256, 256, 0, stream>>>(hs, Wr, br, logits);
  k_router<<<NT_ / 256, 256, 0, stream>>>(logits, wdense, pacc);
  k_aux<<<1, 1, 0, stream>>>(pacc, aux);

  for (int e = 0; e < E_; ++e) {
    dim3 g2(NT_ / 64, H_ / 128);  // (64, 32)
    k_gateup<<<g2, 256, 0, stream>>>(xb, Wg + (size_t)e * H_ * D_,
                                     Wu + (size_t)e * H_ * D_, wdense, e, hb,
                                     NT_, H_, D_);
    dim3 g3(NT_ / 64, D_ / 128);  // (64, 8)
    k_down<<<g3, 256, 0, stream>>>(hb, Wd + (size_t)e * D_ * H_, out, NT_, D_,
                                   H_, e > 0);
  }
}
